// PDRiskNetActorCritic_19481971654887
// MI455X (gfx1250) — matrix operand order repaired and checked by reading, not compiled, not run on hardware
//
#include <hip/hip_runtime.h>
#include <hip/hip_fp16.h>

// ---------------------------------------------------------------------------
// PDRiskNet actor forward, fully fused for gfx1250 (MI455X, wave32, WMMA).
// All GEMMs use v_wmma_f32_16x16x32_f16 (f16 operands, f32 accumulation).
// ---------------------------------------------------------------------------

typedef _Float16 h16;
typedef __attribute__((ext_vector_type(16))) _Float16 v16h;
typedef __attribute__((ext_vector_type(8)))  float    v8f;

union Frag {
  v16h v;
  uint4 q[2];
};

__device__ __forceinline__ v8f wmma16(const Frag& a, const Frag& b, v8f c) {
  return __builtin_amdgcn_wmma_f32_16x16x32_f16(false, a.v, false, b.v,
                                                (short)0, c, false, false);
}

__device__ __forceinline__ float eluf(float x) { return x > 0.f ? x : __expf(x) - 1.f; }
__device__ __forceinline__ float sigm(float x) { return 1.f / (1.f + __expf(-x)); }

// B fragment: 32(K) x 16(N) tile of prepped weight Wp[n][k] (row-major, K contiguous).
// lanes 0-15: K={0..7,16..23} of col n=lane; lanes 16-31: K={8..15,24..31} of col n=lane-16.
__device__ __forceinline__ Frag load_b_frag(const h16* Wp, int Kp, int ncolBase,
                                            int kt, int lane) {
  const h16* p = Wp + (size_t)(ncolBase + (lane & 15)) * Kp + kt * 32 + ((lane >> 4) << 3);
  Frag f;
  f.q[0] = *(const uint4*)p;
  f.q[1] = *(const uint4*)(p + 16);
  return f;
}

// A fragment from global row-major f16 [.., lda], rows = mrow..mrow+15, K tile kbase..+31.
// lanes 0-15: row=lane, K kbase+{0..7,16..23}; lanes 16-31: row=lane-16, K kbase+{8..15,24..31}.
__device__ __forceinline__ Frag load_a_frag_g(const h16* A, size_t lda, int mrow,
                                              int kbase, int lane) {
  const h16* p = A + (size_t)(mrow + (lane & 15)) * lda + kbase + ((lane >> 4) << 3);
  Frag f;
  f.q[0] = *(const uint4*)p;
  f.q[1] = *(const uint4*)(p + 16);
  return f;
}

// ---------------------------------------------------------------------------
// Weight / bias transcoding into WMMA-friendly padded f16 layout.
// src W is [G*H][K] f32 row-major (G gate sections). dst Wp is [G*Hp][Kp] f16,
// zero-padded in both the per-gate N direction (H->Hp) and K (K->Kp).
// ---------------------------------------------------------------------------
__global__ void k_wprep(const float* __restrict__ W, h16* __restrict__ Wp,
                        int K, int Kp, int H, int Hp, int G) {
  int idx = blockIdx.x * blockDim.x + threadIdx.x;
  int total = G * Hp * Kp;
  if (idx >= total) return;
  int k = idx % Kp;
  int j = (idx / Kp) % Hp;
  int g = idx / (Kp * Hp);
  float v = (j < H && k < K) ? W[(size_t)(g * H + j) * K + k] : 0.f;
  Wp[idx] = (h16)v;
}

__global__ void k_bprep(const float* __restrict__ b, float* __restrict__ bp,
                        int H, int Hp, int G) {
  int idx = blockIdx.x * blockDim.x + threadIdx.x;
  if (idx >= G * Hp) return;
  int j = idx % Hp;
  int g = idx / Hp;
  bp[idx] = (j < H) ? b[g * H + j] : 0.f;
}

// x0 concat buffer [2048][320]: cols 0..47 = proprio (f32->f16), rest zeroed
// (mem-GRU kernels later overwrite cols 48..298).
__global__ void k_x0fill(const float* __restrict__ obs, h16* __restrict__ x0) {
  int idx = blockIdx.x * blockDim.x + threadIdx.x;
  if (idx >= 2048 * 320) return;
  int c = idx % 320;
  int b = idx / 320;
  float v = (c < 48) ? obs[(size_t)b * 3120 + c] : 0.f;
  x0[idx] = (h16)v;
}

// ---------------------------------------------------------------------------
// Gather + point encoder: pts[b, idx[t]] -> elu(L1 3->64) -> elu(L2 64->64).
// 256 threads handle 128 rows; layer 1 scalar (3 MACs/out), layer 2 via WMMA.
// ---------------------------------------------------------------------------
__global__ __launch_bounds__(256) void k_encode(
    const float* __restrict__ obs, const int* __restrict__ pidx,
    const float* __restrict__ w1, const float* __restrict__ b1,
    const h16* __restrict__ w2p, const float* __restrict__ b2,
    h16* __restrict__ enc) {
  __shared__ h16 h1[128 * 72];  // stride 72 halves -> conflict-free 16B A-frag reads
  const int tid = threadIdx.x;
  const int rowbase = blockIdx.x * 128;

  {  // phase 1: layer 1 (2 threads per row, 32 feats each)
    int rl = tid >> 1;
    int f0 = (tid & 1) * 32;
    int row = rowbase + rl;
    int b = row >> 9;
    int t = row & 511;
    int p = pidx[t];
    const float* pp = obs + (size_t)b * 3120 + 48 + p * 3;
    float x = pp[0], y = pp[1], z = pp[2];
#pragma unroll 8
    for (int f = f0; f < f0 + 32; ++f) {
      float v = w1[f * 3 + 0] * x + w1[f * 3 + 1] * y + w1[f * 3 + 2] * z + b1[f];
      h1[rl * 72 + f] = (h16)eluf(v);
    }
  }
  __syncthreads();

  // phase 2: layer 2 as 8 waves x (1 m-tile x 4 n-tiles), K=64 -> 2 WMMA/tile
  const int wid = tid >> 5, lane = tid & 31;
  const int mt = wid;
  Frag a0, a1;
  {
    int r = (lane & 15);
    int koff = ((lane >> 4) << 3);
    int o0 = (mt * 16 + r) * 72 + 0 + koff;
    int o1 = (mt * 16 + r) * 72 + 32 + koff;
    a0.q[0] = *(const uint4*)&h1[o0];
    a0.q[1] = *(const uint4*)&h1[o0 + 16];
    a1.q[0] = *(const uint4*)&h1[o1];
    a1.q[1] = *(const uint4*)&h1[o1 + 16];
  }
  const int m0 = mt * 16 + ((lane >> 4) << 3);
#pragma unroll
  for (int nt = 0; nt < 4; ++nt) {
    Frag bb0 = load_b_frag(w2p, 64, nt * 16, 0, lane);
    Frag bb1 = load_b_frag(w2p, 64, nt * 16, 1, lane);
    v8f c = {};
    c = wmma16(a0, bb0, c);
    c = wmma16(a1, bb1, c);
    int col = nt * 16 + (lane & 15);
    float bc = b2[col];
#pragma unroll
    for (int i = 0; i < 8; ++i) {
      enc[(size_t)(rowbase + m0 + i) * 64 + col] = (h16)eluf(c[i] + bc);
    }
  }
}

// ---------------------------------------------------------------------------
// Sequential GRU over T=512. One workgroup owns a 16-row batch slab for the
// whole scan; h lives in ping-pong LDS buffers (f16) + owning wave's f32
// registers. NW waves; wave jt owns hidden-column tile [jt*16, jt*16+16).
// whh/wih B-fragments are loaded once and stay resident for all 512 steps.
// enc A-fragments are double-buffered in registers (prefetch of step t+1
// issued before step t's WMMA), and h uses double-buffered LDS so only one
// barrier per step is required.
// ---------------------------------------------------------------------------
template <int HP, int HSTRIDE, int KTH, int NW>
__global__ __launch_bounds__(NW * 32) void k_gru(
    const h16* __restrict__ enc, const h16* __restrict__ wihp,
    const h16* __restrict__ whhp, const float* __restrict__ bihp,
    const float* __restrict__ bhhp, h16* __restrict__ hout) {
  __shared__ h16 hl[2 * 16 * HSTRIDE];
  const int tid = threadIdx.x, lane = tid & 31, jt = tid >> 5;
  const int b0 = blockIdx.x * 16;

  // zero buffer 0 (initial hidden state)
  for (int i = tid; i < 16 * HSTRIDE; i += NW * 32) hl[i] = (h16)0.f;

  // resident B fragments (whh: 3 gates x KTH k-tiles, wih: 3 gates x 2 k-tiles)
  Frag bwhh[3][KTH];
  Frag bwih[3][2];
#pragma unroll
  for (int g = 0; g < 3; ++g) {
#pragma unroll
    for (int kt = 0; kt < KTH; ++kt)
      bwhh[g][kt] = load_b_frag(whhp, HP, g * HP + jt * 16, kt, lane);
#pragma unroll
    for (int kt = 0; kt < 2; ++kt)
      bwih[g][kt] = load_b_frag(wihp, 64, g * HP + jt * 16, kt, lane);
  }

  const int col = jt * 16 + (lane & 15);
  const float bir = bihp[0 * HP + col], biz = bihp[1 * HP + col], bin = bihp[2 * HP + col];
  const float bhr = bhhp[0 * HP + col], bhz = bhhp[1 * HP + col], bhn = bhhp[2 * HP + col];
  float hreg[8];
#pragma unroll
  for (int i = 0; i < 8; ++i) hreg[i] = 0.f;

  const int mrow_w = ((lane >> 4) << 3);
  const h16* encbase = enc + (size_t)b0 * 512 * 64 + (size_t)(lane & 15) * 512 * 64 +
                       ((lane >> 4) << 3);
  const int a_off = (lane & 15) * HSTRIDE + ((lane >> 4) << 3);

  auto loadenc = [&](Frag(&ae)[2], int t) {
#pragma unroll
    for (int kt = 0; kt < 2; ++kt) {
      const h16* p = encbase + t * 64 + kt * 32;
      ae[kt].q[0] = *(const uint4*)p;
      ae[kt].q[1] = *(const uint4*)(p + 16);
    }
  };

  auto step = [&](const Frag(&ae)[2], int rbuf, int wbuf) {
    v8f cr = {}, cz = {}, cngh = {}, cngx = {};
    const int rb = rbuf * 16 * HSTRIDE;
#pragma unroll
    for (int kt = 0; kt < KTH; ++kt) {
      Frag a;
      int o = rb + a_off + kt * 32;
      a.q[0] = *(const uint4*)&hl[o];
      a.q[1] = *(const uint4*)&hl[o + 16];
      cr = wmma16(a, bwhh[0][kt], cr);
      cz = wmma16(a, bwhh[1][kt], cz);
      cngh = wmma16(a, bwhh[2][kt], cngh);
    }
#pragma unroll
    for (int kt = 0; kt < 2; ++kt) {
      cr = wmma16(ae[kt], bwih[0][kt], cr);
      cz = wmma16(ae[kt], bwih[1][kt], cz);
      cngx = wmma16(ae[kt], bwih[2][kt], cngx);
    }
    const int wb = wbuf * 16 * HSTRIDE;
#pragma unroll
    for (int i = 0; i < 8; ++i) {
      float r = sigm(cr[i] + bir + bhr);
      float z = sigm(cz[i] + biz + bhz);
      float n = tanhf(cngx[i] + bin + r * (cngh[i] + bhn));
      hreg[i] = (1.f - z) * n + z * hreg[i];
      hl[wb + (mrow_w + i) * HSTRIDE + col] = (h16)hreg[i];
    }
    __syncthreads();  // writes to wbuf visible; rbuf reads already complete
  };

  Frag ea[2], eb[2];
  loadenc(ea, 0);
  __syncthreads();  // h buffer 0 zeroed, B frags resident

  for (int t = 0; t < 512; t += 2) {
    loadenc(eb, t + 1);        // prefetch next step's enc while computing
    step(ea, 0, 1);
    if (t + 2 < 512) loadenc(ea, t + 2);
    step(eb, 1, 0);
  }

#pragma unroll
  for (int i = 0; i < 8; ++i)
    hout[(size_t)(b0 + mrow_w + i) * HP + col] = (h16)hreg[i];
}

// ---------------------------------------------------------------------------
// Single-step "mem" GRU with h0 = 0: gh reduces to bhh, so only gx GEMM.
// grid = (M/16, HP/16), one wave per block; writes result into x0 concat.
// ---------------------------------------------------------------------------
template <int HP, int KTH>
__global__ __launch_bounds__(32) void k_memgru(
    const h16* __restrict__ hbuf, const h16* __restrict__ wihp,
    const float* __restrict__ bihp, const float* __restrict__ bhhp,
    h16* __restrict__ x0, int colbase, int Hreal) {
  const int lane = threadIdx.x;
  const int mt = blockIdx.x, jt = blockIdx.y;
  v8f cr = {}, cz = {}, cn = {};
#pragma unroll
  for (int kt = 0; kt < KTH; ++kt) {
    Frag a = load_a_frag_g(hbuf, HP, mt * 16, kt * 32, lane);
    Frag br = load_b_frag(wihp, HP, 0 * HP + jt * 16, kt, lane);
    Frag bz = load_b_frag(wihp, HP, 1 * HP + jt * 16, kt, lane);
    Frag bn = load_b_frag(wihp, HP, 2 * HP + jt * 16, kt, lane);
    cr = wmma16(a, br, cr);
    cz = wmma16(a, bz, cz);
    cn = wmma16(a, bn, cn);
  }
  const int j = jt * 16 + (lane & 15);
  const float bir = bihp[j], biz = bihp[HP + j], bin = bihp[2 * HP + j];
  const float bhr = bhhp[j], bhz = bhhp[HP + j], bhn = bhhp[2 * HP + j];
  const int m0 = mt * 16 + ((lane >> 4) << 3);
#pragma unroll
  for (int i = 0; i < 8; ++i) {
    float r = sigm(cr[i] + bir + bhr);
    float z = sigm(cz[i] + biz + bhz);
    float n = tanhf(cn[i] + bin + r * bhn);
    float h = (1.f - z) * n;
    if (j < Hreal) x0[(size_t)(m0 + i) * 320 + colbase + j] = (h16)h;
  }
}

// ---------------------------------------------------------------------------
// Generic GEMM + bias (+ELU) for the actor MLP. Each wave owns NTPW adjacent
// 16x16 output tiles (independent accumulation chains for ILP, shared A
// fragment). 8 waves per block. ACT=1: elu -> f16 out [M][Np]. ACT=0: f32
// out [M][Nreal] with column bound check (final 12-wide layer).
// ---------------------------------------------------------------------------
template <int ACT, int NTPW>
__global__ __launch_bounds__(256) void k_gemm(
    const h16* __restrict__ A, const h16* __restrict__ B,
    const float* __restrict__ bias, void* __restrict__ outp,
    int M, int Kp, int Np, int Nreal) {
  const int wid = threadIdx.x >> 5, lane = threadIdx.x & 31;
  const int ngrp = (Np >> 4) / NTPW;
  const int grp = blockIdx.x * 8 + wid;
  if (grp >= (M >> 4) * ngrp) return;
  const int mt = grp / ngrp, nt0 = (grp % ngrp) * NTPW;
  v8f zero = {};
  v8f c[NTPW];
#pragma unroll
  for (int j = 0; j < NTPW; ++j) c[j] = zero;
  const int kts = Kp >> 5;
  for (int kt = 0; kt < kts; ++kt) {
    Frag a = load_a_frag_g(A, Kp, mt * 16, kt * 32, lane);
#pragma unroll
    for (int j = 0; j < NTPW; ++j) {
      Frag b = load_b_frag(B, Kp, (nt0 + j) * 16, kt, lane);
      c[j] = wmma16(a, b, c[j]);
    }
  }
  const int m0 = mt * 16 + ((lane >> 4) << 3);
#pragma unroll
  for (int j = 0; j < NTPW; ++j) {
    const int col = (nt0 + j) * 16 + (lane & 15);
    const float bb = bias[col];
    if (ACT) {
      h16* out = (h16*)outp;
#pragma unroll
      for (int i = 0; i < 8; ++i)
        out[(size_t)(m0 + i) * Np + col] = (h16)eluf(c[j][i] + bb);
    } else {
      float* out = (float*)outp;
#pragma unroll
      for (int i = 0; i < 8; ++i)
        if (col < Nreal) out[(size_t)(m0 + i) * Nreal + col] = c[j][i] + bb;
    }
  }
}

// ---------------------------------------------------------------------------
// Host-side orchestration.
// ---------------------------------------------------------------------------
extern "C" void kernel_launch(void* const* d_in, const int* in_sizes, int n_in,
                              void* d_out, int out_size, void* d_ws, size_t ws_size,
                              hipStream_t stream) {
  (void)in_sizes; (void)n_in; (void)out_size; (void)ws_size;

  const float* obs = (const float*)d_in[0];
  const int* prox_idx = (const int*)d_in[1];
  const int* dist_idx = (const int*)d_in[2];
  const float* pe_prox_w1 = (const float*)d_in[3];
  const float* pe_prox_b1 = (const float*)d_in[4];
  const float* pe_prox_w2 = (const float*)d_in[5];
  const float* pe_prox_b2 = (const float*)d_in[6];
  const float* pe_dist_w1 = (const float*)d_in[7];
  const float* pe_dist_b1 = (const float*)d_in[8];
  const float* pe_dist_w2 = (const float*)d_in[9];
  const float* pe_dist_b2 = (const float*)d_in[10];
  const float* gp_wih = (const float*)d_in[11];
  const float* gp_whh = (const float*)d_in[12];
  const float* gp_bih = (const float*)d_in[13];
  const float* gp_bhh = (const float*)d_in[14];
  const float* gd_wih = (const float*)d_in[15];
  const float* gd_whh = (const float*)d_in[16];
  const float* gd_bih = (const float*)d_in[17];
  const float* gd_bhh = (const float*)d_in[18];
  const float* mp_wih = (const float*)d_in[19];
  const float* mp_bih = (const float*)d_in[21];
  const float* mp_bhh = (const float*)d_in[22];
  const float* md_wih = (const float*)d_in[23];
  const float* md_bih = (const float*)d_in[25];
  const float* md_bhh = (const float*)d_in[26];
  const float* aw0 = (const float*)d_in[27];
  const float* ab0 = (const float*)d_in[28];
  const float* aw1 = (const float*)d_in[29];
  const float* ab1 = (const float*)d_in[30];
  const float* aw2 = (const float*)d_in[31];
  const float* ab2 = (const float*)d_in[32];
  const float* aw3 = (const float*)d_in[33];
  const float* ab3 = (const float*)d_in[34];
  const float* aw4 = (const float*)d_in[35];
  const float* ab4 = (const float*)d_in[36];

  char* ws = (char*)d_ws;
  size_t off = 0;
  auto alloc = [&](size_t bytes) -> char* {
    char* p = ws + off;
    off += (bytes + 255) & ~(size_t)255;
    return p;
  };

  // prepped weights (f16) and biases (f32)
  h16* peP_w2p = (h16*)alloc(64 * 64 * 2);
  h16* peD_w2p = (h16*)alloc(64 * 64 * 2);
  h16* gpWihp = (h16*)alloc(576 * 64 * 2);
  h16* gpWhhp = (h16*)alloc(576 * 192 * 2);
  h16* gdWihp = (h16*)alloc(192 * 64 * 2);
  h16* gdWhhp = (h16*)alloc(192 * 64 * 2);
  h16* mpWihp = (h16*)alloc(576 * 192 * 2);
  h16* mdWihp = (h16*)alloc(192 * 64 * 2);
  h16* w0p = (h16*)alloc(1024 * 320 * 2);
  h16* w1p = (h16*)alloc(512 * 1024 * 2);
  h16* w2p = (h16*)alloc(256 * 512 * 2);
  h16* w3p = (h16*)alloc(128 * 256 * 2);
  h16* w4p = (h16*)alloc(16 * 128 * 2);
  float* gpBihp = (float*)alloc(576 * 4);
  float* gpBhhp = (float*)alloc(576 * 4);
  float* mpBihp = (float*)alloc(576 * 4);
  float* mpBhhp = (float*)alloc(576 * 4);
  float* ab4p = (float*)alloc(16 * 4);
  // activations
  h16* encP = (h16*)alloc((size_t)2048 * 512 * 64 * 2);
  h16* encD = (h16*)alloc((size_t)2048 * 512 * 64 * 2);
  h16* hp = (h16*)alloc(2048 * 192 * 2);
  h16* hd = (h16*)alloc(2048 * 64 * 2);
  h16* x0 = (h16*)alloc(2048 * 320 * 2);
  h16* a1 = (h16*)alloc((size_t)2048 * 1024 * 2);
  h16* a2 = (h16*)alloc((size_t)2048 * 512 * 2);
  h16* a3 = (h16*)alloc((size_t)2048 * 256 * 2);
  h16* a4 = (h16*)alloc((size_t)2048 * 128 * 2);

  auto wprep = [&](const float* W, h16* Wp, int K, int Kp, int H, int Hp, int G) {
    int total = G * Hp * Kp;
    k_wprep<<<(total + 255) / 256, 256, 0, stream>>>(W, Wp, K, Kp, H, Hp, G);
  };
  auto bprep = [&](const float* b, float* bp, int H, int Hp, int G) {
    int total = G * Hp;
    k_bprep<<<(total + 255) / 256, 256, 0, stream>>>(b, bp, H, Hp, G);
  };

  // --- stage 0: weight transcode ---
  wprep(pe_prox_w2, peP_w2p, 64, 64, 64, 64, 1);
  wprep(pe_dist_w2, peD_w2p, 64, 64, 64, 64, 1);
  wprep(gp_wih, gpWihp, 64, 64, 187, 192, 3);
  wprep(gp_whh, gpWhhp, 187, 192, 187, 192, 3);
  wprep(gd_wih, gdWihp, 64, 64, 64, 64, 3);
  wprep(gd_whh, gdWhhp, 64, 64, 64, 64, 3);
  wprep(mp_wih, mpWihp, 187, 192, 187, 192, 3);
  wprep(md_wih, mdWihp, 64, 64, 64, 64, 3);
  wprep(aw0, w0p, 299, 320, 1024, 1024, 1);
  wprep(aw1, w1p, 1024, 1024, 512, 512, 1);
  wprep(aw2, w2p, 512, 512, 256, 256, 1);
  wprep(aw3, w3p, 256, 256, 128, 128, 1);
  wprep(aw4, w4p, 128, 128, 12, 16, 1);
  bprep(gp_bih, gpBihp, 187, 192, 3);
  bprep(gp_bhh, gpBhhp, 187, 192, 3);
  bprep(mp_bih, mpBihp, 187, 192, 3);
  bprep(mp_bhh, mpBhhp, 187, 192, 3);
  bprep(ab4, ab4p, 12, 16, 1);

  // --- stage 1: x0 proprio + zero pad ---
  k_x0fill<<<(2048 * 320 + 255) / 256, 256, 0, stream>>>(obs, x0);

  // --- stage 2: gather + point encoders ---
  k_encode<<<8192, 256, 0, stream>>>(obs, prox_idx, pe_prox_w1, pe_prox_b1,
                                     peP_w2p, pe_prox_b2, encP);
  k_encode<<<8192, 256, 0, stream>>>(obs, dist_idx, pe_dist_w1, pe_dist_b1,
                                     peD_w2p, pe_dist_b2, encD);

  // --- stage 3: sequential GRUs (512 steps) ---
  k_gru<192, 200, 6, 12><<<128, 12 * 32, 0, stream>>>(encP, gpWihp, gpWhhp,
                                                      gpBihp, gpBhhp, hp);
  k_gru<64, 72, 2, 4><<<128, 4 * 32, 0, stream>>>(encD, gdWihp, gdWhhp,
                                                  gd_bih, gd_bhh, hd);

  // --- stage 4: mem GRUs (single step, h0=0) into x0 concat ---
  k_memgru<192, 6><<<dim3(128, 12), 32, 0, stream>>>(hp, mpWihp, mpBihp, mpBhhp,
                                                     x0, 48, 187);
  k_memgru<64, 2><<<dim3(128, 4), 32, 0, stream>>>(hd, mdWihp, md_bih, md_bhh,
                                                   x0, 235, 64);

  // --- stage 5: actor MLP ---
  k_gemm<1, 2><<<512, 256, 0, stream>>>(x0, w0p, ab0, a1, 2048, 320, 1024, 1024);
  k_gemm<1, 2><<<256, 256, 0, stream>>>(a1, w1p, ab1, a2, 2048, 1024, 512, 512);
  k_gemm<1, 2><<<128, 256, 0, stream>>>(a2, w2p, ab2, a3, 2048, 512, 256, 256);
  k_gemm<1, 2><<<64, 256, 0, stream>>>(a3, w3p, ab3, a4, 2048, 256, 128, 128);
  k_gemm<0, 1><<<16, 256, 0, stream>>>(a4, w4p, ab4p, d_out, 2048, 128, 16, 12);
}
